// DenseMultiRangeAttention_22935125360698
// MI455X (gfx1250) — matrix operand order repaired and checked by reading, not compiled, not run on hardware
//
#include <hip/hip_runtime.h>
#include <hip/hip_bf16.h>
#include <math.h>

typedef __attribute__((ext_vector_type(16))) _Float16 v16h;
typedef __attribute__((ext_vector_type(8)))  _Float16 v8h;
typedef __attribute__((ext_vector_type(8)))  float    v8f;
typedef int v4i_vs __attribute__((vector_size(16)));   // matches builtin diag type

#define EMBED_DIM 96
#define NUM_HEADS 4
#define HEAD_DIM  24
#define HH 64
#define WW 64
#define HWPIX 4096              // 64*64
#define BATCH 2
#define NTOT  (BATCH * HWPIX)   // 8192 GEMM columns
#define CCAT  288               // 96 * 3 ranges
#define MT    6                 // 96/16 M-tiles, all owned by one wave

#define GAS __attribute__((address_space(1)))
#define LAS __attribute__((address_space(3)))

// ---------------------------------------------------------------------------
// f32 -> f16 elementwise (weights)
// ---------------------------------------------------------------------------
__global__ void cvt_f32_to_f16(const float* __restrict__ src,
                               _Float16* __restrict__ dst, int n) {
    int i = blockIdx.x * blockDim.x + threadIdx.x;
    if (i < n) dst[i] = (_Float16)src[i];
}

// ---------------------------------------------------------------------------
// x (B, C, HW) f32  ->  xT (B*HW, C) f16   (row j = b*HW + p)
// ---------------------------------------------------------------------------
__global__ void cvt_x_transpose(const float* __restrict__ x,
                                _Float16* __restrict__ xt, int n) {
    int i = blockIdx.x * blockDim.x + threadIdx.x;
    if (i >= n) return;
    int b = i / (EMBED_DIM * HWPIX);
    int r = i - b * (EMBED_DIM * HWPIX);
    int c = r / HWPIX;
    int p = r - c * HWPIX;
    xt[((size_t)b * HWPIX + p) * EMBED_DIM + c] = (_Float16)x[i];
}

// ---------------------------------------------------------------------------
// WMMA GEMM + bias, LDS-staged weights:
//   Out[b][m][p] = sum_k Wh[m][k] * XT[b*HW+p][k] + bias[m]
//   Wh : (96 x Kd) f16 row-major  -> staged once per WG into LDS via
//        global_load_async_to_lds_b128 (fallback: regular load + ds_store)
//   XT : (N x Kd) f16 row-major   (B operand, pre-transposed, 32B loads)
//   Out: (B, 96, HW) f32
// Workgroup = 8 waves = 96x128 output tile; each wave owns a 16-column
// stripe and all 6 M-tiles -> 6 v_wmma per unique B fragment, A fragments
// from LDS (ds_load_b128).
// ---------------------------------------------------------------------------
__global__ void __launch_bounds__(256)
wmma_gemm_bias(const _Float16* __restrict__ Wh,
               const _Float16* __restrict__ XT,
               const float* __restrict__ bias,
               float* __restrict__ Out,
               int Kd) {
    __shared__ __align__(16) _Float16 sW[EMBED_DIM * CCAT]; // worst case 55296 B

    const int tid = threadIdx.x;

    // ---- stage weight matrix (96 x Kd f16) into LDS, 16B chunks ----
    const int chunks = (EMBED_DIM * Kd) >> 3;   // # of 16-byte chunks
    for (int idx = tid; idx < chunks; idx += 256) {
#if defined(__has_builtin) && __has_builtin(__builtin_amdgcn_global_load_async_to_lds_b128)
        __builtin_amdgcn_global_load_async_to_lds_b128(
            (v4i_vs*)(void*)(const void*)(Wh + (size_t)idx * 8),
            (LAS v4i_vs*)(LAS void*)(void*)(&sW[(size_t)idx * 8]),
            /*offset=*/0, /*cpol=*/0);
#else
        *(v8h*)(&sW[(size_t)idx * 8]) = *(const v8h*)(Wh + (size_t)idx * 8);
#endif
    }
#if defined(__has_builtin) && __has_builtin(__builtin_amdgcn_global_load_async_to_lds_b128)
#if __has_builtin(__builtin_amdgcn_s_wait_asynccnt)
    __builtin_amdgcn_s_wait_asynccnt(0);
#else
    asm volatile("s_wait_asynccnt 0" ::: "memory");
#endif
#endif
    __syncthreads();

    // ---- per-wave compute ----
    const int wave = tid >> 5;
    const int lane = tid & 31;
    const int half = lane >> 4;       // 0: lanes 0-15, 1: lanes 16-31
    const int l16  = lane & 15;
    const int bcol = blockIdx.x * 128 + wave * 16 + l16;   // B column of lane

    v8f acc[MT] = {};
    const _Float16* __restrict__ xcol = XT + (size_t)bcol * Kd;

    for (int kb = 0; kb < Kd; kb += 32) {
        // B fragment (32x16 f16): lane holds its column, K kb+half*16+{0..15}
        v16h bf = *(const v16h*)(xcol + kb + half * 16);
        if (kb + 32 < Kd) __builtin_prefetch(xcol + kb + 32, 0, 0);

#pragma unroll
        for (int mt = 0; mt < MT; ++mt) {
            // A fragment (16x32 f16) from LDS: row mt*16+l16,
            // elems 0-15 = K kb+half*16+{0..15}
            const _Float16* wrow = &sW[(mt * 16 + l16) * Kd + kb + half * 16];
            v8h lo = *(const v8h*)(wrow);
            v8h hi = *(const v8h*)(wrow + 8);
            v16h af;
#pragma unroll
            for (int i = 0; i < 8; ++i) { af[i] = lo[i]; af[8 + i] = hi[i]; }

            acc[mt] = __builtin_amdgcn_wmma_f32_16x16x32_f16(
                          /*neg_a=*/false, af, /*neg_b=*/false, bf,
                          /*c_mod=*/(short)0, acc[mt],
                          /*reuse_a=*/false, /*reuse_b=*/false);
        }
    }

    const int bb = bcol >> 12;        // batch = column / 4096
    const int p  = bcol & 4095;
#pragma unroll
    for (int mt = 0; mt < MT; ++mt) {
#pragma unroll
        for (int v = 0; v < 8; ++v) {
            int r = mt * 16 + v + half * 8;
            Out[((size_t)bb * EMBED_DIM + r) * HWPIX + p] = acc[mt][v] + bias[r];
        }
    }
}

// ---------------------------------------------------------------------------
// Multi-range local attention. One thread per (batch, head, pixel).
// Online softmax over k*k neighbors; zero-padded neighbors contribute
// score==0 (dot with zero K) to the softmax and zero V (matches reference).
// Writes concat directly as f16 in (B*HW, 288) transposed layout with
// channel = head*72 + range*24 + d.
// ---------------------------------------------------------------------------
__global__ void multi_range_attn(const float* __restrict__ Q,
                                 const float* __restrict__ K,
                                 const float* __restrict__ V,
                                 _Float16* __restrict__ catT) {
    const int total = BATCH * NUM_HEADS * HWPIX;
    int tid = blockIdx.x * blockDim.x + threadIdx.x;
    if (tid >= total) return;

    int b    = tid / (NUM_HEADS * HWPIX);
    int rem  = tid - b * (NUM_HEADS * HWPIX);
    int head = rem / HWPIX;
    int p    = rem - head * HWPIX;
    int y = p >> 6;
    int x = p & 63;

    const size_t base = ((size_t)b * EMBED_DIM + head * HEAD_DIM) * HWPIX;
    const float inv_scale = 0.2041241452319315f;   // 1/sqrt(24)

    float q[HEAD_DIM];
#pragma unroll
    for (int d = 0; d < HEAD_DIM; ++d)
        q[d] = Q[base + (size_t)d * HWPIX + p];

    _Float16* outp = catT + ((size_t)b * HWPIX + p) * CCAT + head * 72;

    const int ksz[3] = {7, 9, 11};
    for (int ri = 0; ri < 3; ++ri) {
        int kk  = ksz[ri];
        int pad = kk >> 1;
        float m = -INFINITY, s = 0.f;
        float acc[HEAD_DIM];
#pragma unroll
        for (int d = 0; d < HEAD_DIM; ++d) acc[d] = 0.f;

        for (int i = 0; i < kk; ++i) {
            int yy = y + i - pad;
            for (int j = 0; j < kk; ++j) {
                int xx = x + j - pad;
                bool inb = (yy >= 0) & (yy < HH) & (xx >= 0) & (xx < WW);
                int np = yy * WW + xx;
                float sc = 0.f;
                if (inb) {
                    float dot = 0.f;
#pragma unroll
                    for (int d = 0; d < HEAD_DIM; ++d)
                        dot += q[d] * K[base + (size_t)d * HWPIX + np];
                    sc = dot * inv_scale;
                }
                float mn = fmaxf(m, sc);
                float f  = __expf(m - mn);
                float e  = __expf(sc - mn);
                s = s * f + e;
#pragma unroll
                for (int d = 0; d < HEAD_DIM; ++d) acc[d] *= f;
                if (inb) {
#pragma unroll
                    for (int d = 0; d < HEAD_DIM; ++d)
                        acc[d] += e * V[base + (size_t)d * HWPIX + np];
                }
                m = mn;
            }
        }
        float inv_s = 1.f / s;
#pragma unroll
        for (int d = 0; d < HEAD_DIM; ++d)
            outp[ri * HEAD_DIM + d] = (_Float16)(acc[d] * inv_s);
    }
}

// ---------------------------------------------------------------------------
extern "C" void kernel_launch(void* const* d_in, const int* in_sizes, int n_in,
                              void* d_out, int out_size, void* d_ws, size_t ws_size,
                              hipStream_t stream) {
    const float* x  = (const float*)d_in[0];
    const float* wq = (const float*)d_in[1];
    const float* bq = (const float*)d_in[2];
    const float* wk = (const float*)d_in[3];
    const float* bk = (const float*)d_in[4];
    const float* wv = (const float*)d_in[5];
    const float* bv = (const float*)d_in[6];
    const float* wo = (const float*)d_in[7];
    const float* bo = (const float*)d_in[8];
    float* out = (float*)d_out;

    // workspace carve-up (all offsets 256-B aligned)
    char* ws = (char*)d_ws;
    const size_t qkvBytes = (size_t)BATCH * EMBED_DIM * HWPIX * sizeof(float); // 3 MiB each
    float*    Qf  = (float*)(ws);
    float*    Kf  = (float*)(ws + qkvBytes);
    float*    Vf  = (float*)(ws + 2 * qkvBytes);
    _Float16* xt  = (_Float16*)(ws + 3 * qkvBytes);                       // (N, 96)
    _Float16* cat = (_Float16*)(ws + 3 * qkvBytes
                                   + (size_t)NTOT * EMBED_DIM * 2);        // (N, 288)
    char* wbase = ws + 3 * qkvBytes + (size_t)NTOT * EMBED_DIM * 2
                     + (size_t)NTOT * CCAT * 2;
    _Float16* wqh = (_Float16*)(wbase);
    _Float16* wkh = (_Float16*)(wbase + 18432);
    _Float16* wvh = (_Float16*)(wbase + 2 * 18432);
    _Float16* woh = (_Float16*)(wbase + 3 * 18432);

    // 1) weight + activation conversion
    cvt_f32_to_f16<<<(EMBED_DIM * EMBED_DIM + 255) / 256, 256, 0, stream>>>(wq, wqh, EMBED_DIM * EMBED_DIM);
    cvt_f32_to_f16<<<(EMBED_DIM * EMBED_DIM + 255) / 256, 256, 0, stream>>>(wk, wkh, EMBED_DIM * EMBED_DIM);
    cvt_f32_to_f16<<<(EMBED_DIM * EMBED_DIM + 255) / 256, 256, 0, stream>>>(wv, wvh, EMBED_DIM * EMBED_DIM);
    cvt_f32_to_f16<<<(EMBED_DIM * CCAT + 255) / 256, 256, 0, stream>>>(wo, woh, EMBED_DIM * CCAT);
    {
        int n = BATCH * EMBED_DIM * HWPIX;
        cvt_x_transpose<<<(n + 255) / 256, 256, 0, stream>>>(x, xt, n);
    }

    // 2) Q/K/V projections: 96x128 tile per WG -> 8192/128 = 64 blocks
    wmma_gemm_bias<<<NTOT / 128, 256, 0, stream>>>(wqh, xt, bq, Qf, EMBED_DIM);
    wmma_gemm_bias<<<NTOT / 128, 256, 0, stream>>>(wkh, xt, bk, Kf, EMBED_DIM);
    wmma_gemm_bias<<<NTOT / 128, 256, 0, stream>>>(wvh, xt, bv, Vf, EMBED_DIM);

    // 3) multi-range attention -> concat (f16, transposed)
    {
        int n = BATCH * NUM_HEADS * HWPIX;                 // 32768 threads
        multi_range_attn<<<(n + 255) / 256, 256, 0, stream>>>(Qf, Kf, Vf, cat);
    }

    // 4) output projection: Wo (96x288) x concat^T -> d_out (B,96,64,64)
    wmma_gemm_bias<<<NTOT / 128, 256, 0, stream>>>(woh, cat, bo, out, CCAT);
}
